// QuanvNet_17325898072363
// MI455X (gfx1250) — compile-verified, matrix operand order checked by reading, not run on hardware
//
#include <hip/hip_runtime.h>
#include <hip/hip_bf16.h>
#include <math.h>

// CDNA5 wave32 WMMA types
typedef __attribute__((ext_vector_type(16))) _Float16 v16h;
typedef __attribute__((ext_vector_type(8)))  float    v8f;

// K index inside a 16x32 f16 fragment for element e (0..15) of this lane.
// Per ISA 7.12.2: lanes 0-15 hold K {0..7,16..23}, lanes 16-31 hold K {8..15,24..31}.
__device__ __forceinline__ int frag_k(int e, int kb) {
    return (e < 8) ? (kb + e) : (16 + kb + (e - 8));
}

__device__ __forceinline__ float wave_reduce_add(float v) {
    #pragma unroll
    for (int m = 16; m > 0; m >>= 1) v += __shfl_xor(v, m, 32);
    return v;
}

// ---------------------------------------------------------------------------
// Pack a row-major f32 matrix [rows x K] into WMMA fragment-ordered f16:
// dst[((tile*nsteps + s)*32 + lane)*16 + e] = src[row*K + k],
//   row = tile*16 + (lane&15), kb = lane<16?0:8, k = s*32 + frag_k(e,kb).
// Zero-pads k >= K. Used for conv1_w (A), conv2_w (A), fc1_w (B^T == same).
// ---------------------------------------------------------------------------
__global__ __launch_bounds__(256)
void pack_frag_kernel(const float* __restrict__ src, _Float16* __restrict__ dst,
                      int K, int nsteps, int ntiles)
{
    int total = ntiles * nsteps * 32 * 16;
    for (int t = blockIdx.x * blockDim.x + threadIdx.x; t < total;
         t += gridDim.x * blockDim.x) {
        int e    = t & 15;
        int lane = (t >> 4) & 31;
        int rem  = t >> 9;
        int s    = rem % nsteps;
        int tile = rem / nsteps;
        int kb   = (lane < 16) ? 0 : 8;
        int row  = tile * 16 + (lane & 15);
        int k    = s * 32 + frag_k(e, kb);
        dst[t] = (_Float16)((k < K) ? src[row * K + k] : 0.f);
    }
}

// ---------------------------------------------------------------------------
// Kernel A: fused pre-convs (1->4 s2, 4->4 s2, 4->4 s1) + closed-form quanv.
// quanv: d_w = cos(theta_w)cos(x) - sin(theta_w)sin(phi_w)sin(x), ez = cumprod.
// ---------------------------------------------------------------------------
__global__ __launch_bounds__(256)
void quanv_pre_kernel(const float* __restrict__ x,
                      const float* __restrict__ w1, const float* __restrict__ b1,
                      const float* __restrict__ w2, const float* __restrict__ b2,
                      const float* __restrict__ w3, const float* __restrict__ b3,
                      const float* __restrict__ qw,
                      float* __restrict__ qout)
{
    __shared__ float sx [64 * 64];
    __shared__ float sh1[4 * 32 * 32];
    __shared__ float sh2[4 * 16 * 16];
    __shared__ float sh3[4 * 14 * 14];
    __shared__ float qc[8];

    const int b   = blockIdx.x;
    const int tid = threadIdx.x;

    for (int t = tid; t < 64 * 64; t += 256) sx[t] = x[b * 4096 + t];
    if (tid < 4) {
        float phi = qw[tid * 3 + 0], th = qw[tid * 3 + 1];
        qc[tid]     = cosf(th);
        qc[4 + tid] = sinf(th) * sinf(phi);
    }
    __syncthreads();

    for (int t = tid; t < 4 * 32 * 32; t += 256) {           // conv1 s2 p1
        int oc = t >> 10, oy = (t >> 5) & 31, ox = t & 31;
        float s = b1[oc];
        #pragma unroll
        for (int ky = 0; ky < 3; ++ky) {
            int y = 2 * oy - 1 + ky;
            if (y < 0 || y > 63) continue;
            #pragma unroll
            for (int kx = 0; kx < 3; ++kx) {
                int xx = 2 * ox - 1 + kx;
                if (xx < 0 || xx > 63) continue;
                s += sx[y * 64 + xx] * w1[oc * 9 + ky * 3 + kx];
            }
        }
        sh1[t] = fmaxf(s, 0.f);
    }
    __syncthreads();

    for (int t = tid; t < 4 * 16 * 16; t += 256) {           // conv2 s2 p1
        int oc = t >> 8, oy = (t >> 4) & 15, ox = t & 15;
        float s = b2[oc];
        for (int ic = 0; ic < 4; ++ic) {
            const float* wp = &w2[(oc * 4 + ic) * 9];
            const float* hp = &sh1[ic * 1024];
            #pragma unroll
            for (int ky = 0; ky < 3; ++ky) {
                int y = 2 * oy - 1 + ky;
                if (y < 0 || y > 31) continue;
                #pragma unroll
                for (int kx = 0; kx < 3; ++kx) {
                    int xx = 2 * ox - 1 + kx;
                    if (xx < 0 || xx > 31) continue;
                    s += hp[y * 32 + xx] * wp[ky * 3 + kx];
                }
            }
        }
        sh2[t] = fmaxf(s, 0.f);
    }
    __syncthreads();

    for (int t = tid; t < 4 * 14 * 14; t += 256) {           // conv3 s1 p0
        int oc = t / 196, r = t % 196, oy = r / 14, ox = r % 14;
        float s = b3[oc];
        for (int ic = 0; ic < 4; ++ic) {
            const float* wp = &w3[(oc * 4 + ic) * 9];
            const float* hp = &sh2[ic * 256];
            #pragma unroll
            for (int ky = 0; ky < 3; ++ky)
                #pragma unroll
                for (int kx = 0; kx < 3; ++kx)
                    s += hp[(oy + ky) * 16 + (ox + kx)] * wp[ky * 3 + kx];
        }
        sh3[t] = s;
    }
    __syncthreads();

    for (int t = tid; t < 4 * 7 * 7; t += 256) {             // quanv closed form
        int c = t / 49, r = t % 49, i = r / 7, j = r % 7;
        const float* hp = &sh3[c * 196];
        float d[4];
        #pragma unroll
        for (int w = 0; w < 4; ++w) {
            int kh = w >> 1, kw = w & 1;
            float v = hp[(2 * i + kh) * 14 + (2 * j + kw)];
            d[w] = qc[w] * cosf(v) - qc[4 + w] * sinf(v);
        }
        float ez = 1.f;
        float* qo = &qout[(b * 16 + c * 4) * 49 + r];
        #pragma unroll
        for (int w = 0; w < 4; ++w) { ez *= d[w]; qo[w * 49] = ez; }
    }
}

// ---------------------------------------------------------------------------
// Kernel B: per-image conv1(16->32)+GN+ReLU, conv2(32->64)+GN+ReLU, maxpool.
// Im2col is built directly in WMMA fragment order -> B-frags are 2 LDS b128s;
// weight A-frags are contiguous 32B global loads of prepacked f16 panels.
// ---------------------------------------------------------------------------
__global__ __launch_bounds__(256)
void conv_gn_kernel(const float* __restrict__ qbuf,
                    const v16h*  __restrict__ wf1,   // packed conv1_w  [2][5][32]
                    const float* __restrict__ c1b,
                    const float* __restrict__ g1w, const float* __restrict__ g1b,
                    const v16h*  __restrict__ wf2,   // packed conv2_w  [4][9][32]
                    const float* __restrict__ c2b,
                    const float* __restrict__ g2w, const float* __restrict__ g2b,
                    float* __restrict__ pooled)
{
    __shared__ float D1[32 * 64];          //  8 KB conv1 out (N 49->64)
    __shared__ float D2[64 * 64];          // 16 KB conv2 out
    __shared__ v16h  ImF[9 * 4 * 32];      // 36 KB im2col frags [(s*4+nt)*32+lane]
    __shared__ float red[16];

    const int b    = blockIdx.x;
    const int tid  = threadIdx.x;
    const int lane = tid & 31;
    const int wv   = tid >> 5;
    const int mhb  = (lane < 16) ? 0 : 8;
    const int ln   = lane & 15;
    _Float16* ImH  = (_Float16*)ImF;

    // ---- im2col frags for conv1: K=144 (pad 160 = 5 steps) x N=49 (pad 64)
    const float* qi = &qbuf[b * 16 * 49];
    for (int t = tid; t < 5 * 4 * 32 * 16; t += 256) {
        int e = t & 15, L = (t >> 4) & 31, r = t >> 9;
        int nt = r & 3, s = r >> 2;
        int kb = (L < 16) ? 0 : 8;
        int n = nt * 16 + (L & 15);
        int k = s * 32 + frag_k(e, kb);
        float v = 0.f;
        if (k < 144 && n < 49) {
            int c = k / 9, rr = k % 9, kh = rr / 3, kw = rr % 3;
            int i = n / 7, j = n % 7;
            int y = i + kh - 1, xx = j + kw - 1;
            if (y >= 0 && y < 7 && xx >= 0 && xx < 7) v = qi[c * 49 + y * 7 + xx];
        }
        ImH[t] = (_Float16)v;
    }
    __syncthreads();

    // ---- conv1 GEMM: M=32, N=64, K=160; tile (mt,nt) per wave
    {
        int mt = wv >> 2, nt = wv & 3;
        v8f acc = {0.f, 0.f, 0.f, 0.f, 0.f, 0.f, 0.f, 0.f};
        for (int s = 0; s < 5; ++s) {
            v16h av = wf1[(mt * 5 + s) * 32 + lane];
            v16h bv = ImF[(s * 4 + nt) * 32 + lane];
            acc = __builtin_amdgcn_wmma_f32_16x16x32_f16(false, av, false, bv,
                                                         (short)0, acc, false, false);
        }
        int n = nt * 16 + ln;
        #pragma unroll
        for (int r = 0; r < 8; ++r) {
            int mrow = mt * 16 + mhb + r;
            D1[mrow * 64 + n] = acc[r] + c1b[mrow];
        }
    }
    __syncthreads();

    // ---- GN1 (8 groups x 4ch x 49px) + ReLU: wave wv handles group wv
    {
        float s = 0.f, ss = 0.f;
        for (int t = lane; t < 4 * 49; t += 32) {
            int c = wv * 4 + t / 49, n = t % 49;
            float v = D1[c * 64 + n]; s += v; ss += v * v;
        }
        s = wave_reduce_add(s); ss = wave_reduce_add(ss);
        if (lane == 0) {
            float mu = s / 196.f, var = ss / 196.f - mu * mu;
            red[wv] = mu; red[8 + wv] = rsqrtf(var + 1e-5f);
        }
    }
    __syncthreads();
    for (int t = tid; t < 32 * 64; t += 256) {
        int c = t >> 6, n = t & 63;
        float v = 0.f;
        if (n < 49) {
            int g = c >> 2;
            v = fmaxf((D1[t] - red[g]) * red[8 + g] * g1w[c] + g1b[c], 0.f);
        }
        D1[t] = v;
    }
    __syncthreads();

    // ---- im2col frags for conv2: K=288 (9 steps) x N=64, src = D1
    for (int t = tid; t < 9 * 4 * 32 * 16; t += 256) {
        int e = t & 15, L = (t >> 4) & 31, r = t >> 9;
        int nt = r & 3, s = r >> 2;
        int kb = (L < 16) ? 0 : 8;
        int n = nt * 16 + (L & 15);
        int k = s * 32 + frag_k(e, kb);
        float v = 0.f;
        if (n < 49) {
            int c = k / 9, rr = k % 9, kh = rr / 3, kw = rr % 3;
            int i = n / 7, j = n % 7;
            int y = i + kh - 1, xx = j + kw - 1;
            if (y >= 0 && y < 7 && xx >= 0 && xx < 7) v = D1[c * 64 + y * 7 + xx];
        }
        ImH[t] = (_Float16)v;
    }
    __syncthreads();

    // ---- conv2 GEMM: M=64, N=64, K=288; 16 tiles, 2 per wave
    for (int tt = 0; tt < 2; ++tt) {
        int T = wv + tt * 8;
        int mt = T >> 2, nt = T & 3;
        v8f acc = {0.f, 0.f, 0.f, 0.f, 0.f, 0.f, 0.f, 0.f};
        for (int s = 0; s < 9; ++s) {
            v16h av = wf2[(mt * 9 + s) * 32 + lane];
            v16h bv = ImF[(s * 4 + nt) * 32 + lane];
            acc = __builtin_amdgcn_wmma_f32_16x16x32_f16(false, av, false, bv,
                                                         (short)0, acc, false, false);
        }
        int n = nt * 16 + ln;
        #pragma unroll
        for (int r = 0; r < 8; ++r) {
            int mrow = mt * 16 + mhb + r;
            D2[mrow * 64 + n] = acc[r] + c2b[mrow];
        }
    }
    __syncthreads();

    // ---- GN2 (8 groups x 8ch x 49px) + ReLU
    {
        float s = 0.f, ss = 0.f;
        for (int t = lane; t < 8 * 49; t += 32) {
            int c = wv * 8 + t / 49, n = t % 49;
            float v = D2[c * 64 + n]; s += v; ss += v * v;
        }
        s = wave_reduce_add(s); ss = wave_reduce_add(ss);
        if (lane == 0) {
            float mu = s / 392.f, var = ss / 392.f - mu * mu;
            red[wv] = mu; red[8 + wv] = rsqrtf(var + 1e-5f);
        }
    }
    __syncthreads();
    for (int t = tid; t < 64 * 64; t += 256) {
        int c = t >> 6, n = t & 63;
        if (n < 49) {
            int g = c >> 3;
            D2[t] = fmaxf((D2[t] - red[g]) * red[8 + g] * g2w[c] + g2b[c], 0.f);
        }
    }
    __syncthreads();

    // ---- maxpool 2x2 (7x7 -> 3x3), flatten (c, pi, pj)
    for (int t = tid; t < 64 * 9; t += 256) {
        int c = t / 9, r = t % 9, pi = r / 3, pj = r % 3;
        const float* hp = &D2[c * 64];
        float m0 = fmaxf(hp[(2 * pi) * 7 + 2 * pj],     hp[(2 * pi) * 7 + 2 * pj + 1]);
        float m1 = fmaxf(hp[(2 * pi + 1) * 7 + 2 * pj], hp[(2 * pi + 1) * 7 + 2 * pj + 1]);
        pooled[b * 576 + t] = fmaxf(m0, m1);
    }
}

// ---------------------------------------------------------------------------
// Kernel C: fc1 (576->128, WMMA, ReLU) + fc2 (128->10). 16 batch rows / block.
// Activations staged in LDS in fragment order; fc1_w prepacked in global.
// ---------------------------------------------------------------------------
__global__ __launch_bounds__(256)
void fc_kernel(const float* __restrict__ pooled,
               const v16h*  __restrict__ wfc,   // packed fc1_w [8][18][32]
               const float* __restrict__ b1,
               const float* __restrict__ w2, const float* __restrict__ b2,
               float* __restrict__ out, int Btot)
{
    __shared__ v16h  Xf[18 * 32];     // 18 KB activation frags [s*32+lane]
    __shared__ float Y [16 * 128];    //  8 KB fc1 output

    const int blk  = blockIdx.x;
    const int tid  = threadIdx.x;
    const int lane = tid & 31;
    const int wv   = tid >> 5;
    const int mhb  = (lane < 16) ? 0 : 8;
    const int ln   = lane & 15;
    _Float16* XfH  = (_Float16*)Xf;

    // Build A-frags: row = L&15 (batch row in block), k = s*32 + frag_k(e,kb)
    for (int t = tid; t < 18 * 32 * 16; t += 256) {
        int e = t & 15, L = (t >> 4) & 31, s = t >> 9;
        int kb = (L < 16) ? 0 : 8;
        int row = L & 15;
        int k = s * 32 + frag_k(e, kb);
        int gm = blk * 16 + row;
        XfH[t] = (_Float16)((gm < Btot) ? pooled[gm * 576 + k] : 0.f);
    }
    __syncthreads();

    // fc1: M=16 (batch) x N=128 x K=576; 8 N-tiles, one per wave
    {
        int n = wv * 16 + ln;
        v8f acc = {0.f, 0.f, 0.f, 0.f, 0.f, 0.f, 0.f, 0.f};
        for (int s = 0; s < 18; ++s) {
            v16h av = Xf[s * 32 + lane];
            v16h bv = wfc[(wv * 18 + s) * 32 + lane];
            acc = __builtin_amdgcn_wmma_f32_16x16x32_f16(false, av, false, bv,
                                                         (short)0, acc, false, false);
        }
        #pragma unroll
        for (int r = 0; r < 8; ++r)
            Y[(mhb + r) * 128 + n] = fmaxf(acc[r] + b1[n], 0.f);
    }
    __syncthreads();

    // fc2: 16 rows x 10 outputs, dot over 128
    if (tid < 160) {
        int m = tid / 10, o = tid % 10;
        int gm = blk * 16 + m;
        if (gm < Btot) {
            float s = b2[o];
            for (int k = 0; k < 128; ++k) s += Y[m * 128 + k] * w2[o * 128 + k];
            out[gm * 10 + o] = s;
        }
    }
}

// ---------------------------------------------------------------------------
extern "C" void kernel_launch(void* const* d_in, const int* in_sizes, int n_in,
                              void* d_out, int out_size, void* d_ws, size_t ws_size,
                              hipStream_t stream) {
    const float* x      = (const float*)d_in[0];
    const float* pre_w1 = (const float*)d_in[1];
    const float* pre_b1 = (const float*)d_in[2];
    const float* pre_w2 = (const float*)d_in[3];
    const float* pre_b2 = (const float*)d_in[4];
    const float* pre_w3 = (const float*)d_in[5];
    const float* pre_b3 = (const float*)d_in[6];
    const float* qw     = (const float*)d_in[7];
    const float* c1w    = (const float*)d_in[8];
    const float* c1b    = (const float*)d_in[9];
    const float* g1w    = (const float*)d_in[10];
    const float* g1b    = (const float*)d_in[11];
    const float* c2w    = (const float*)d_in[12];
    const float* c2b    = (const float*)d_in[13];
    const float* g2w    = (const float*)d_in[14];
    const float* g2b    = (const float*)d_in[15];
    const float* fw1    = (const float*)d_in[16];
    const float* fb1    = (const float*)d_in[17];
    const float* fw2    = (const float*)d_in[18];
    const float* fb2    = (const float*)d_in[19];

    const int Btot = in_sizes[0] / (64 * 64);

    // Workspace layout (all 32B aligned)
    char*  ws     = (char*)d_ws;
    float* qbuf   = (float*)ws;                         // B*16*49 f32
    size_t off    = (size_t)Btot * 16 * 49 * sizeof(float);
    float* pooled = (float*)(ws + off);                 // B*576 f32
    off          += (size_t)Btot * 576 * sizeof(float);
    _Float16* wf1 = (_Float16*)(ws + off);              // 2*5*32*16 f16 = 10 KB
    off          += 2 * 5 * 32 * 16 * sizeof(_Float16);
    _Float16* wf2 = (_Float16*)(ws + off);              // 4*9*32*16 f16 = 36 KB
    off          += 4 * 9 * 32 * 16 * sizeof(_Float16);
    _Float16* wfc = (_Float16*)(ws + off);              // 8*18*32*16 f16 = 144 KB

    // Pack weight panels into fragment order (cheap; amortized over B blocks)
    pack_frag_kernel<<<20,  256, 0, stream>>>(c1w, wf1, 144,  5, 2);
    pack_frag_kernel<<<72,  256, 0, stream>>>(c2w, wf2, 288,  9, 4);
    pack_frag_kernel<<<288, 256, 0, stream>>>(fw1, wfc, 576, 18, 8);

    quanv_pre_kernel<<<Btot, 256, 0, stream>>>(x, pre_w1, pre_b1, pre_w2, pre_b2,
                                               pre_w3, pre_b3, qw, qbuf);
    conv_gn_kernel<<<Btot, 256, 0, stream>>>(qbuf, (const v16h*)wf1, c1b, g1w, g1b,
                                             (const v16h*)wf2, c2b, g2w, g2b, pooled);
    fc_kernel<<<(Btot + 15) / 16, 256, 0, stream>>>(pooled, (const v16h*)wfc,
                                                    fb1, fw2, fb2,
                                                    (float*)d_out, Btot);
}
